// MixtureOfBidders_27839978013048
// MI455X (gfx1250) — compile-verified, matrix operand order checked by reading, not run on hardware
//
#include <hip/hip_runtime.h>
#include <cstdint>

#define T_TOK 4096
#define DDIM  1024
#define IDIM  2048
#define NEXP  8
#define KT1   (DDIM / 32)   // 32 k-tiles in GEMM1
#define KT2   (IDIM / 32)   // 64 k-tiles in GEMM2
#define ROWS_CAP 9216       // 8192 bucket rows + per-expert pad to 128

typedef __attribute__((ext_vector_type(16))) __bf16 v16bf;
typedef __attribute__((ext_vector_type(8)))  float  v8f;
typedef __attribute__((ext_vector_type(4)))  unsigned int v4u;
typedef __attribute__((ext_vector_type(8)))  int v8i;
typedef __attribute__((ext_vector_type(4)))  int v4i;

#if __has_builtin(__builtin_amdgcn_tensor_load_to_lds) && __has_builtin(__builtin_amdgcn_s_wait_tensorcnt)
#define HAVE_TDM 1
#else
#define HAVE_TDM 0
#endif

// ---------- helpers ----------
__device__ __forceinline__ unsigned short f2bf(float f) {
    union { float f; uint32_t u; } v; v.f = f;
    uint32_t r = v.u + 0x7FFFu + ((v.u >> 16) & 1u);   // round-to-nearest-even
    return (unsigned short)(r >> 16);
}
__device__ __forceinline__ uint32_t packbf(float a, float b) {
    return (uint32_t)f2bf(a) | ((uint32_t)f2bf(b) << 16);
}
__device__ __forceinline__ v8f wmma_bf16(v16bf a, v16bf b, v8f c) {
    return __builtin_amdgcn_wmma_f32_16x16x32_bf16(
        false, a, false, b, (short)0, c, false, false);
}
__device__ __forceinline__ v8f v8f_zero() {
    v8f z;
#pragma unroll
    for (int j = 0; j < 8; ++j) z[j] = 0.f;
    return z;
}
// 16-bit WMMA operand layout (ISA 7.12.2): lanes 0-15 hold K {0..7,16..23},
// lanes 16-31 hold K {8..15,24..31}; element index = (K&7) + (K>=16 ? 8 : 0).
__device__ __forceinline__ int frag_lane(int mn, int kk) {
    return (mn & 15) + (((kk >> 3) & 1) << 4);
}
__device__ __forceinline__ int frag_idx(int kk) {
    return (kk & 7) + ((kk & 16) ? 8 : 0);
}
__device__ __forceinline__ uint32_t lds_off(const void* p) {
    return (uint32_t)(uintptr_t)p;   // addr[31:0] of a shared pointer = LDS offset
}

#if HAVE_TDM
// One 2-D TDM descriptor: 8 rows x 1KB, row stride = stride8 (8-byte units),
// packed contiguously into LDS at ldsaddr.
__device__ __forceinline__ void tdm_load_2d(const void* gptr, uint32_t ldsaddr,
                                            uint32_t stride8) {
    uint64_t ga = (uint64_t)(uintptr_t)gptr;
    v4u g0;
    g0[0] = 1u;                                                  // count=1
    g0[1] = ldsaddr;                                             // lds_addr
    g0[2] = (uint32_t)ga;                                        // global_addr lo
    g0[3] = (uint32_t)((ga >> 32) & 0x01FFFFFFu) | (2u << 30);   // addr[56:32]|type=2
    v8i g1;
    g1[0] = 3 << 16;                                 // data_size = 8B
    g1[1] = (int)((stride8 & 0xFFFFu) << 16);        // tensor_dim0[15:0]
    g1[2] = (int)((stride8 >> 16) | (8u << 16));     // tensor_dim0[31:16]|tensor_dim1=8
    g1[3] = (int)(128u << 16);                       // tile_dim0 = 128 (8B) = 1KB
    g1[4] = 8;                                       // tile_dim1 = 8
    g1[5] = (int)stride8;                            // tensor_dim0_stride lo
    g1[6] = 0;
    g1[7] = 0;
    v4i z4 = {0, 0, 0, 0};
#if defined(__clang_major__) && __clang_major__ >= 23
    v8i z8;
#pragma unroll
    for (int i = 0; i < 8; ++i) z8[i] = 0;
    __builtin_amdgcn_tensor_load_to_lds(g0, g1, z4, z4, z8, 0);
#else
    __builtin_amdgcn_tensor_load_to_lds(g0, g1, z4, z4, 0);
#endif
}
#endif

// Fallback: 8 chunks x 1KB, cooperative b128 copy by 256 threads.
__device__ __forceinline__ void copy_chunks(void* dst, const void* src,
                                            size_t strideBytes, int tid) {
    char* d = (char*)dst;
    const char* s = (const char*)src;
    for (int i = tid; i < 512; i += 256) {
        int c = i >> 6;
        int o = (i & 63) << 4;
        *(uint4*)(d + c * 1024 + o) = *(const uint4*)(s + (size_t)c * strideBytes + o);
    }
}

// ---------- 1) routing ----------
__global__ __launch_bounds__(256) void k_route(const float* __restrict__ x,
                                               const float* __restrict__ Wc,
                                               const float* __restrict__ bc,
                                               int* __restrict__ tokidx,
                                               float* __restrict__ wts) {
    __shared__ float sWc[NEXP * DDIM];
    const int tid = threadIdx.x;
    for (int i = tid; i < NEXP * DDIM; i += 256) sWc[i] = Wc[i];
    __syncthreads();
    const int lane = tid & 31, w = tid >> 5;
    const int t = blockIdx.x * 8 + w;
    float acc[NEXP];
#pragma unroll
    for (int e = 0; e < NEXP; ++e) acc[e] = 0.f;
    const float* xp = x + (size_t)t * DDIM;
    for (int i = 0; i < DDIM / 32; ++i) {
        float xv = xp[i * 32 + lane];
#pragma unroll
        for (int e = 0; e < NEXP; ++e)
            acc[e] = fmaf(xv, sWc[e * DDIM + i * 32 + lane], acc[e]);
    }
#pragma unroll
    for (int e = 0; e < NEXP; ++e) {
        acc[e] += __shfl_xor(acc[e], 16, 32);
        acc[e] += __shfl_xor(acc[e], 8, 32);
        acc[e] += __shfl_xor(acc[e], 4, 32);
        acc[e] += __shfl_xor(acc[e], 2, 32);
        acc[e] += __shfl_xor(acc[e], 1, 32);
    }
    if (lane == 0) {
        float conf[NEXP];
#pragma unroll
        for (int e = 0; e < NEXP; ++e)
            conf[e] = 1.f / (1.f + __expf(-(acc[e] + bc[e])));
        int i0 = 0; float v0 = conf[0];
#pragma unroll
        for (int e = 1; e < NEXP; ++e) if (conf[e] > v0) { v0 = conf[e]; i0 = e; }
        int i1 = (i0 == 0) ? 1 : 0; float v1 = conf[i1];
#pragma unroll
        for (int e = 0; e < NEXP; ++e)
            if (e != i0 && conf[e] > v1) { v1 = conf[e]; i1 = e; }
        float e1  = __expf(v1 - v0);
        float inv = 1.f / (1.f + e1);
        tokidx[2 * t] = i0; tokidx[2 * t + 1] = i1;
        wts[2 * t] = inv;   wts[2 * t + 1] = e1 * inv;
    }
}

// ---------- 2) per-expert token lists ----------
__global__ void k_bucket(const int* __restrict__ tokidx,
                         uint32_t* __restrict__ bucket,
                         uint32_t* __restrict__ counts) {
    const int e = blockIdx.x;
    const int lane = threadIdx.x;
    uint32_t base = 0;
    for (int c = 0; c < T_TOK / 32; ++c) {
        int t = c * 32 + lane;
        int i0 = tokidx[2 * t], i1 = tokidx[2 * t + 1];
        bool match = (i0 == e) || (i1 == e);
        uint32_t slot = (i0 == e) ? 0u : 1u;
        uint32_t mask = (uint32_t)__ballot(match);
        if (match) {
            uint32_t off = __popc(mask & ((1u << lane) - 1u));
            bucket[e * T_TOK + base + off] = (uint32_t)t | (slot << 16);
        }
        base += __popc(mask);
    }
    if (lane == 0) counts[e] = base;
}

// ---------- 3) 128-row-padded exclusive prefix ----------
__global__ void k_prefix(const uint32_t* __restrict__ counts,
                         uint32_t* __restrict__ pbase) {
    if (threadIdx.x == 0 && blockIdx.x == 0) {
        uint32_t b = 0;
        for (int e = 0; e < NEXP; ++e) {
            pbase[e] = b;
            b += (counts[e] + 127u) & ~127u;
        }
    }
}

// ---------- 4a) gather + convert + WMMA-swizzle X rows ----------
__global__ __launch_bounds__(256) void k_packA(const float* __restrict__ x,
                                               const uint32_t* __restrict__ bucket,
                                               const uint32_t* __restrict__ counts,
                                               const uint32_t* __restrict__ pbase,
                                               unsigned short* __restrict__ xswz) {
    const int e = blockIdx.y;
    const int rt = blockIdx.x;
    const int n_e = (int)counts[e];
    if (rt * 16 >= n_e) return;
    __shared__ int stok[16];
    const int tid = threadIdx.x;
    if (tid < 16) {
        int r = rt * 16 + tid;
        stok[tid] = (int)(bucket[e * T_TOK + ((r < n_e) ? r : 0)] & 0xFFFFu);
    }
    __syncthreads();
    const size_t rtG = (pbase[e] >> 4) + rt;
    for (int i = tid; i < 16 * (DDIM / 2); i += 256) {
        int m = i >> 9;                       // DDIM/2 = 512 pairs per row
        int k = (i & 511) << 1;
        const float* s = x + (size_t)stok[m] * DDIM + k;
        int kt = k >> 5, kk = k & 31;
        size_t di = ((rtG * KT1 + kt) * 32 + frag_lane(m, kk)) * 8 + (frag_idx(kk) >> 1);
        ((uint32_t*)xswz)[di] = packbf(s[0], s[1]);
    }
}

// ---------- 4b) convert + WMMA-swizzle a weight matrix [E][K][N] ----------
__global__ __launch_bounds__(256) void k_packW(const float* __restrict__ src,
                                               unsigned short* __restrict__ dst,
                                               int K, int N) {
    const int e = blockIdx.y;
    const int ntile = blockIdx.x;
    const int tid = threadIdx.x;
    const int KT = K >> 5;
    for (int i = tid; i < 8 * K; i += 256) {   // 16 n x K/2 k-pairs
        int n16 = i & 15;
        int k = (i >> 4) << 1;
        const float* s = src + (size_t)e * K * N + (size_t)k * N + (size_t)ntile * 16 + n16;
        int kt = k >> 5, kk = k & 31;
        size_t di = ((((size_t)e * (N >> 4) + ntile) * KT + kt) * 32
                     + frag_lane(n16, kk)) * 8 + (frag_idx(kk) >> 1);
        ((uint32_t*)dst)[di] = packbf(s[0], s[N]);
    }
}

// ---------- 5) GEMM1: h = silu(X@Wg)*(X@Wu); TDM-staged, ping-pong ----------
__global__ __launch_bounds__(256) void k_gemm1(const unsigned short* __restrict__ xswz,
                                               const unsigned short* __restrict__ wgsw,
                                               const unsigned short* __restrict__ wusw,
                                               const uint32_t* __restrict__ counts,
                                               const uint32_t* __restrict__ pbase,
                                               unsigned short* __restrict__ hswz) {
    const int e   = blockIdx.z;
    const int n_e = (int)counts[e];
    const int row0 = blockIdx.y * 128;
    if (row0 >= n_e) return;
    const int col0 = blockIdx.x * 128;

    __shared__ __align__(32) unsigned short lA [2][4096];
    __shared__ __align__(32) unsigned short lBg[2][4096];
    __shared__ __align__(32) unsigned short lBu[2][4096];

    const int tid = threadIdx.x, lane = tid & 31, w = tid >> 5;
    const int wr = w >> 1, wc = w & 1;                  // 4x2 waves, 32x64 each
    const size_t rt0 = (pbase[e] >> 4) + (row0 >> 4);   // 8 consecutive row tiles
    const size_t nb0 = (size_t)e * (IDIM >> 4) + (col0 >> 4);

    auto issue = [&](int kt, int p) {
#if HAVE_TDM
        if (w == 0)
            tdm_load_2d((const char*)xswz + (rt0 * KT1 + kt) * 1024,
                        lds_off(&lA[p][0]), KT1 * 128);
        else if (w == 1)
            tdm_load_2d((const char*)wgsw + (nb0 * KT1 + kt) * 1024,
                        lds_off(&lBg[p][0]), KT1 * 128);
        else if (w == 2)
            tdm_load_2d((const char*)wusw + (nb0 * KT1 + kt) * 1024,
                        lds_off(&lBu[p][0]), KT1 * 128);
#else
        copy_chunks(&lA [p][0], (const char*)xswz + (rt0 * KT1 + kt) * 1024,
                    (size_t)KT1 * 1024, tid);
        copy_chunks(&lBg[p][0], (const char*)wgsw + (nb0 * KT1 + kt) * 1024,
                    (size_t)KT1 * 1024, tid);
        copy_chunks(&lBu[p][0], (const char*)wusw + (nb0 * KT1 + kt) * 1024,
                    (size_t)KT1 * 1024, tid);
#endif
    };

    v8f cg[2][4], cu[2][4];
#pragma unroll
    for (int i = 0; i < 2; ++i)
#pragma unroll
        for (int j = 0; j < 4; ++j) { cg[i][j] = v8f_zero(); cu[i][j] = v8f_zero(); }

    issue(0, 0);
    for (int kt = 0; kt < KT1; ++kt) {
        const int p = kt & 1;
#if HAVE_TDM
        if (w <= 2) __builtin_amdgcn_s_wait_tensorcnt(0);
#endif
        __syncthreads();
        if (kt + 1 < KT1) issue(kt + 1, p ^ 1);

        v16bf a0 = *(const v16bf*)&lA[p][((wr * 2 + 0) * 32 + lane) * 16];
        v16bf a1 = *(const v16bf*)&lA[p][((wr * 2 + 1) * 32 + lane) * 16];
#pragma unroll
        for (int nt = 0; nt < 4; ++nt) {
            v16bf bg = *(const v16bf*)&lBg[p][((wc * 4 + nt) * 32 + lane) * 16];
            v16bf bu = *(const v16bf*)&lBu[p][((wc * 4 + nt) * 32 + lane) * 16];
            cg[0][nt] = wmma_bf16(a0, bg, cg[0][nt]);
            cg[1][nt] = wmma_bf16(a1, bg, cg[1][nt]);
            cu[0][nt] = wmma_bf16(a0, bu, cu[0][nt]);
            cu[1][nt] = wmma_bf16(a1, bu, cu[1][nt]);
        }
    }
    // epilogue: SwiGLU, write h directly in GEMM2's A-fragment layout
    const int nloc = lane & 15;
    const int mhi  = (lane >> 4) << 3;
    const size_t pb = pbase[e] >> 4;
#pragma unroll
    for (int mt = 0; mt < 2; ++mt)
#pragma unroll
        for (int nt = 0; nt < 4; ++nt)
#pragma unroll
            for (int j = 0; j < 8; ++j) {
                int m = wr * 32 + mt * 16 + mhi + j;
                int r = row0 + m;
                if (r < n_e) {
                    float g = cg[mt][nt][j];
                    float u = cu[mt][nt][j];
                    float h = (g / (1.f + __expf(-g))) * u;
                    int col = col0 + wc * 64 + nt * 16 + nloc;
                    int kt2 = col >> 5, kk = col & 31;
                    size_t di = (((pb + (r >> 4)) * KT2 + kt2) * 32
                                 + frag_lane(r, kk)) * 16 + frag_idx(kk);
                    hswz[di] = f2bf(h);
                }
            }
}

// ---------- 6) GEMM2: y = h@Wd; scatter w*y into per-slot planes ----------
__global__ __launch_bounds__(256) void k_gemm2(const unsigned short* __restrict__ hswz,
                                               const unsigned short* __restrict__ wdsw,
                                               const uint32_t* __restrict__ bucket,
                                               const uint32_t* __restrict__ counts,
                                               const uint32_t* __restrict__ pbase,
                                               const float* __restrict__ wts,
                                               float* __restrict__ contrib) {
    const int e   = blockIdx.z;
    const int n_e = (int)counts[e];
    const int row0 = blockIdx.y * 128;
    if (row0 >= n_e) return;
    const int col0 = blockIdx.x * 128;

    __shared__ __align__(32) unsigned short lA[2][4096];
    __shared__ __align__(32) unsigned short lB[2][4096];

    const int tid = threadIdx.x, lane = tid & 31, w = tid >> 5;
    const int wr = w >> 1, wc = w & 1;
    const size_t rt0 = (pbase[e] >> 4) + (row0 >> 4);
    const size_t nb0 = (size_t)e * (DDIM >> 4) + (col0 >> 4);

    auto issue = [&](int kt, int p) {
#if HAVE_TDM
        if (w == 0)
            tdm_load_2d((const char*)hswz + (rt0 * KT2 + kt) * 1024,
                        lds_off(&lA[p][0]), KT2 * 128);
        else if (w == 1)
            tdm_load_2d((const char*)wdsw + (nb0 * KT2 + kt) * 1024,
                        lds_off(&lB[p][0]), KT2 * 128);
#else
        copy_chunks(&lA[p][0], (const char*)hswz + (rt0 * KT2 + kt) * 1024,
                    (size_t)KT2 * 1024, tid);
        copy_chunks(&lB[p][0], (const char*)wdsw + (nb0 * KT2 + kt) * 1024,
                    (size_t)KT2 * 1024, tid);
#endif
    };

    v8f cc[2][4];
#pragma unroll
    for (int i = 0; i < 2; ++i)
#pragma unroll
        for (int j = 0; j < 4; ++j) cc[i][j] = v8f_zero();

    issue(0, 0);
    for (int kt = 0; kt < KT2; ++kt) {
        const int p = kt & 1;
#if HAVE_TDM
        if (w <= 1) __builtin_amdgcn_s_wait_tensorcnt(0);
#endif
        __syncthreads();
        if (kt + 1 < KT2) issue(kt + 1, p ^ 1);

        v16bf a0 = *(const v16bf*)&lA[p][((wr * 2 + 0) * 32 + lane) * 16];
        v16bf a1 = *(const v16bf*)&lA[p][((wr * 2 + 1) * 32 + lane) * 16];
#pragma unroll
        for (int nt = 0; nt < 4; ++nt) {
            v16bf b = *(const v16bf*)&lB[p][((wc * 4 + nt) * 32 + lane) * 16];
            cc[0][nt] = wmma_bf16(a0, b, cc[0][nt]);
            cc[1][nt] = wmma_bf16(a1, b, cc[1][nt]);
        }
    }
    const int nloc = lane & 15;
    const int mhi  = (lane >> 4) << 3;
#pragma unroll
    for (int mt = 0; mt < 2; ++mt)
#pragma unroll
        for (int nt = 0; nt < 4; ++nt)
#pragma unroll
            for (int j = 0; j < 8; ++j) {
                int m = wr * 32 + mt * 16 + mhi + j;
                int r = row0 + m;
                if (r < n_e) {
                    uint32_t en = bucket[e * T_TOK + r];
                    uint32_t tok = en & 0xFFFFu;
                    uint32_t slot = en >> 16;
                    float wgt = wts[tok * 2 + slot];
                    int col = col0 + wc * 64 + nt * 16 + nloc;
                    contrib[(size_t)slot * T_TOK * DDIM + (size_t)tok * DDIM + col] =
                        wgt * cc[mt][nt][j];
                }
            }
}

// ---------- 7) out = contrib0 + contrib1 ----------
__global__ void k_combine(const float* __restrict__ contrib, float* __restrict__ out) {
    const size_t n = (size_t)T_TOK * DDIM;
    for (size_t i = (size_t)blockIdx.x * 256 + threadIdx.x; i < n;
         i += (size_t)gridDim.x * 256)
        out[i] = contrib[i] + contrib[n + i];
}

// ---------- launcher ----------
extern "C" void kernel_launch(void* const* d_in, const int* /*in_sizes*/, int /*n_in*/,
                              void* d_out, int /*out_size*/, void* d_ws, size_t /*ws_size*/,
                              hipStream_t stream) {
    const float* x  = (const float*)d_in[0];
    const float* Wc = (const float*)d_in[1];
    const float* bc = (const float*)d_in[2];
    const float* Wg = (const float*)d_in[3];
    const float* Wu = (const float*)d_in[4];
    const float* Wd = (const float*)d_in[5];
    float* out = (float*)d_out;

    char* ws = (char*)d_ws;
    size_t off = 0;
    unsigned short* xswz = (unsigned short*)(ws + off); off += (size_t)ROWS_CAP * DDIM * 2;
    unsigned short* hswz = (unsigned short*)(ws + off); off += (size_t)ROWS_CAP * IDIM * 2;
    unsigned short* wgsw = (unsigned short*)(ws + off); off += (size_t)NEXP * DDIM * IDIM * 2;
    unsigned short* wusw = (unsigned short*)(ws + off); off += (size_t)NEXP * DDIM * IDIM * 2;
    unsigned short* wdsw = (unsigned short*)(ws + off); off += (size_t)NEXP * IDIM * DDIM * 2;
    float* contrib = (float*)(ws + off);                off += (size_t)2 * T_TOK * DDIM * 4;
    float* wts = (float*)(ws + off);                    off += (size_t)T_TOK * 2 * 4;
    int* tokidx = (int*)(ws + off);                     off += (size_t)T_TOK * 2 * 4;
    uint32_t* bucket = (uint32_t*)(ws + off);           off += (size_t)NEXP * T_TOK * 4;
    uint32_t* counts = (uint32_t*)(ws + off);           off += 256;
    uint32_t* pbase  = (uint32_t*)(ws + off);           off += 256;

    k_route <<<T_TOK / 8, 256, 0, stream>>>(x, Wc, bc, tokidx, wts);
    k_bucket<<<NEXP, 32, 0, stream>>>(tokidx, bucket, counts);
    k_prefix<<<1, 1, 0, stream>>>(counts, pbase);
    k_packW <<<dim3(IDIM / 16, NEXP), 256, 0, stream>>>(Wg, wgsw, DDIM, IDIM);
    k_packW <<<dim3(IDIM / 16, NEXP), 256, 0, stream>>>(Wu, wusw, DDIM, IDIM);
    k_packW <<<dim3(DDIM / 16, NEXP), 256, 0, stream>>>(Wd, wdsw, IDIM, DDIM);
    k_packA <<<dim3(T_TOK / 16, NEXP), 256, 0, stream>>>(x, bucket, counts, pbase, xswz);
    k_gemm1 <<<dim3(IDIM / 128, T_TOK / 128, NEXP), 256, 0, stream>>>(
        xswz, wgsw, wusw, counts, pbase, hswz);
    k_gemm2 <<<dim3(DDIM / 128, T_TOK / 128, NEXP), 256, 0, stream>>>(
        hswz, wdsw, bucket, counts, pbase, wts, contrib);
    k_combine<<<2048, 256, 0, stream>>>(contrib, out);
}
